// PerceiverSelfAttention_12824772346505
// MI455X (gfx1250) — matrix operand order repaired and checked by reading, not compiled, not run on hardware
//
#include <hip/hip_runtime.h>

typedef __attribute__((ext_vector_type(16))) __bf16 v16bf;
typedef __attribute__((ext_vector_type(8)))  float  v8f;
typedef unsigned int v4u __attribute__((ext_vector_type(4)));
typedef int          v4i __attribute__((ext_vector_type(4)));
typedef int          v8i __attribute__((ext_vector_type(8)));

#if __has_include(<hip/amd_detail/amd_gfx1250_TDM.h>)
#define CDNA5_TDM_6ARG 1
#endif

// ---------------------------------------------------------------- helpers
__device__ __forceinline__ v16bf frag2(const __bf16* p0, const __bf16* p1) {
  v16bf r;
  ((uint4*)&r)[0] = *(const uint4*)p0;   // elements 0..7
  ((uint4*)&r)[1] = *(const uint4*)p1;   // elements 8..15
  return r;
}

// TDM: async-load a 2D bf16 tile [tile_h x tile_w] (row stride = stride_elems)
// from global to LDS, padding each 256B row chunk with 16B so the LDS row
// stride becomes tile_w + 8 bf16 (for tile_w = 128 -> stride 136).
// D# layout per CDNA5 ISA §8.3/§8.4. Issue from ONE wave only.
__device__ __forceinline__ void tdm_load_tile_bf16(unsigned int lds_byte_off,
                                                   const void* gptr,
                                                   unsigned int tile_w,
                                                   unsigned int tile_h,
                                                   unsigned int stride_elems) {
  unsigned long long ga = (unsigned long long)(uintptr_t)gptr;
  v4u g0;
  g0.x = 1u;                                            // count=1 (valid user D#)
  g0.y = lds_byte_off;                                  // lds_addr
  g0.z = (unsigned int)(ga & 0xffffffffu);              // global_addr[31:0]
  g0.w = (unsigned int)((ga >> 32) & 0x1ffffffu)        // global_addr[56:32]
       | (2u << 30);                                    // type=2 ("image")
  const unsigned int d0 = (1u << 16)   // data_size = 2 bytes
                        | (1u << 20)   // pad_enable
                        | (5u << 22)   // pad_interval: 64 DWORDs (256B)
                        | (3u << 25);  // pad_amount:   4 DWORDs (16B)
  const unsigned int d1 = (tile_w & 0xffffu) << 16;                    // tensor_dim0 lo
  const unsigned int d2 = (tile_w >> 16) | ((tile_h & 0xffffu) << 16); // dim0 hi, dim1 lo
  const unsigned int d3 = (tile_h >> 16) | (tile_w << 16);             // dim1 hi, tile_dim0
  const unsigned int d4 = tile_h & 0xffffu;                            // tile_dim1 (tile_dim2=0)
  const unsigned int d5 = stride_elems;                                // tensor_dim0_stride lo
  v8i g1 = { (int)d0, (int)d1, (int)d2, (int)d3, (int)d4, (int)d5, 0, 0 };
  v4i gz = { 0, 0, 0, 0 };
#if defined(CDNA5_TDM_6ARG)
  v8i gz8 = { 0, 0, 0, 0, 0, 0, 0, 0 };
  __builtin_amdgcn_tensor_load_to_lds(g0, g1, gz, gz, gz8, 0);
#else
  __builtin_amdgcn_tensor_load_to_lds(g0, g1, gz, gz, 0);
#endif
}

// ---------------------------------------------------------------- LayerNorm + cast to bf16
__global__ __launch_bounds__(256)
void ln_cast_kernel(const float* __restrict__ x, const float* __restrict__ w,
                    const float* __restrict__ bvec, __bf16* __restrict__ y, int dim) {
  const int row = blockIdx.x;
  const int tid = threadIdx.x;
  const float* xr = x + (size_t)row * dim;
  float s = 0.f, s2 = 0.f;
  for (int i = tid; i < dim; i += 256) { float v = xr[i]; s += v; s2 += v * v; }
#pragma unroll
  for (int off = 16; off; off >>= 1) { s += __shfl_xor(s, off, 32); s2 += __shfl_xor(s2, off, 32); }
  __shared__ float sm[8], sm2[8];
  const int wid = tid >> 5;
  if ((tid & 31) == 0) { sm[wid] = s; sm2[wid] = s2; }
  __syncthreads();
  if (wid == 0) {
    float a  = (tid < 8) ? sm[tid]  : 0.f;
    float a2 = (tid < 8) ? sm2[tid] : 0.f;
#pragma unroll
    for (int off = 4; off; off >>= 1) { a += __shfl_xor(a, off, 32); a2 += __shfl_xor(a2, off, 32); }
    if (tid == 0) { sm[0] = a; sm2[0] = a2; }
  }
  __syncthreads();
  const float mean = sm[0] / (float)dim;
  const float var  = sm2[0] / (float)dim - mean * mean;
  const float inv  = rsqrtf(var + 1e-5f);
  __bf16* yr = y + (size_t)row * dim;
  for (int i = tid; i < dim; i += 256) {
    float v = (xr[i] - mean) * inv * w[i] + bvec[i];
    yr[i] = (__bf16)v;
  }
}

// ---------------------------------------------------------------- f32 -> bf16 cast
__global__ __launch_bounds__(256)
void cast_bf16_kernel(const float* __restrict__ x, __bf16* __restrict__ y, int n) {
  int i = blockIdx.x * 256 + threadIdx.x;
  if (i < n) y[i] = (__bf16)x[i];
}

// ---------------------------------------------------------------- bf16 WMMA GEMM: Y = X @ W + bias
// Compile-time KDIM/N so all tile addressing folds to immediate offsets.
// BM=64, BN=128, BK=32; 256 threads = 8 waves, each wave 32x32 (2x2 wmma tiles)
template <int KDIM, int N>
__global__ __launch_bounds__(256)
void gemm_bf16_kernel(const __bf16* __restrict__ X, const __bf16* __restrict__ W,
                      const float* __restrict__ bias, __bf16* __restrict__ Y) {
  __shared__ __bf16 As[64 * 40];    // [row][k], stride 40
  __shared__ __bf16 Bt[128 * 40];   // W transposed: [n][k], stride 40
  const int tid  = threadIdx.x;
  const int lane = tid & 31, wid = tid >> 5;
  const int m = lane & 15, half = lane >> 4;
  const int wm = wid >> 2, wn = wid & 3;
  const int bm = blockIdx.y * 64, bn = blockIdx.x * 128;

  v8f cacc[2][2];
  v8f vz = {};
#pragma unroll
  for (int i = 0; i < 2; ++i)
#pragma unroll
    for (int j = 0; j < 2; ++j) cacc[i][j] = vz;

  const int arow = tid >> 2, acol = (tid & 3) * 8;   // A: 64 rows x 32 cols
  const int tk = tid >> 4,  tn8 = (tid & 15) * 8;    // W: 16 k-rows x (16x8) n per pass

  for (int k0 = 0; k0 < KDIM; k0 += 32) {
    *(uint4*)(As + arow * 40 + acol) =
        *(const uint4*)(X + (size_t)(bm + arow) * KDIM + k0 + acol);
#pragma unroll
    for (int p = 0; p < 2; ++p) {
      const int k = k0 + p * 16 + tk;
      uint4 wv = *(const uint4*)(W + (size_t)k * N + bn + tn8);
      const __bf16* we = (const __bf16*)&wv;
#pragma unroll
      for (int e = 0; e < 8; ++e) Bt[(tn8 + e) * 40 + p * 16 + tk] = we[e];
    }
    __syncthreads();

    v16bf a[2], bb[2];
#pragma unroll
    for (int t = 0; t < 2; ++t) {
      const __bf16* pa = As + (wm * 32 + t * 16 + m) * 40;
      a[t] = frag2(pa + 16 * half, pa + 16 * half + 8);
      const __bf16* pb = Bt + (wn * 32 + t * 16 + m) * 40 + 16 * half;
      bb[t] = frag2(pb, pb + 8);
    }
#pragma unroll
    for (int i = 0; i < 2; ++i)
#pragma unroll
      for (int j = 0; j < 2; ++j)
        cacc[i][j] = __builtin_amdgcn_wmma_f32_16x16x32_bf16(
            false, a[i], false, bb[j], (short)0, cacc[i][j], false, false);
    __syncthreads();
  }

  // epilogue: base pointer once, constant offsets fold into store immediates
  const int row0 = bm + wm * 32 + 8 * half;
  const int col0 = bn + wn * 32 + m;
  __bf16* yb = Y + (size_t)row0 * N + col0;
  const float bias0 = bias[col0];
  const float bias1 = bias[col0 + 16];
#pragma unroll
  for (int i = 0; i < 2; ++i)
#pragma unroll
    for (int r = 0; r < 8; ++r) {
      yb[(i * 16 + r) * N]      = (__bf16)(cacc[i][0][r] + bias0);
      yb[(i * 16 + r) * N + 16] = (__bf16)(cacc[i][1][r] + bias1);
    }
}

// ---------------------------------------------------------------- fused flash attention
// Q[B,MQ,H*128], K/V[B,MK,H*128] bf16; mask[B,MK] int; out[B,MQ,H*128] f32
// 1 block = (b, h, 64-query tile); 128 threads = 4 waves, wave w owns 16 query rows.
// Q and K tiles staged by the Tensor Data Mover (async, TENSORcnt), V staged
// transposed by VALU/DS (TDM cannot transpose).
__global__ __launch_bounds__(128)
void flash_attn_kernel(const __bf16* __restrict__ Q, const __bf16* __restrict__ K,
                       const __bf16* __restrict__ V, const int* __restrict__ mask,
                       float* __restrict__ out) {
  constexpr int MQc = 512, MKc = 2048, Dc = 128, HD = 1024;
  const float scale = 0.08838834764831845f;  // 1/sqrt(128)

  const int blk = blockIdx.x;
  const int qt = blk & 7, h = (blk >> 3) & 7, b = blk >> 6;
  const int q0 = qt * 64;
  const int tid = threadIdx.x;
  const int lane = tid & 31, w = tid >> 5;
  const int m = lane & 15, half = lane >> 4;

  __shared__ __bf16 Qs[64 * 136];
  __shared__ __bf16 Ks[64 * 136];
  __shared__ __bf16 Vt[128 * 72];   // V transposed: [d][key]
  __shared__ __bf16 Pb[64 * 72];    // probs bf16: [q][key]
  __shared__ int    smask[64];

  // stage Q tile (64 x 128) via TDM from wave 0
  if (w == 0) {
    tdm_load_tile_bf16((unsigned int)(uintptr_t)(void*)Qs,
                       Q + ((size_t)(b * MQc + q0)) * HD + h * Dc, Dc, 64, HD);
    __builtin_amdgcn_s_wait_tensorcnt(0);
  }
  __syncthreads();

  v16bf qf[4];
#pragma unroll
  for (int kb = 0; kb < 4; ++kb) {
    const __bf16* base = Qs + (w * 16 + m) * 136 + kb * 32;
    qf[kb] = frag2(base + 16 * half, base + 16 * half + 8);
  }

  float mrow[8], lrow[8];
  v8f acc[8];
  v8f vz = {};
#pragma unroll
  for (int r = 0; r < 8; ++r) { mrow[r] = -3.0e38f; lrow[r] = 0.f; }
#pragma unroll
  for (int nt = 0; nt < 8; ++nt) acc[nt] = vz;

  for (int kb0 = 0; kb0 < MKc / 64; ++kb0) {
    const int kv0 = kb0 * 64;
    if (kb0 + 1 < MKc / 64)
      __builtin_prefetch(V + ((size_t)(b * MKc + kv0 + 64)) * HD + h * Dc, 0, 1);

    // K tile via TDM (wave 0), V transposed manually (all threads)
    if (w == 0)
      tdm_load_tile_bf16((unsigned int)(uintptr_t)(void*)Ks,
                         K + ((size_t)(b * MKc + kv0)) * HD + h * Dc, Dc, 64, HD);
#pragma unroll
    for (int j = 0; j < 8; ++j) {
      const int c = tid + j * 128;
      const int row = c >> 4, c8 = (c & 15) * 8;
      uint4 vv = *(const uint4*)(V + ((size_t)(b * MKc + kv0 + row)) * HD + h * Dc + c8);
      const __bf16* ve = (const __bf16*)&vv;
#pragma unroll
      for (int e = 0; e < 8; ++e) Vt[(c8 + e) * 72 + row] = ve[e];
    }
    if (tid < 64) smask[tid] = mask[b * MKc + kv0 + tid];
    if (w == 0) __builtin_amdgcn_s_wait_tensorcnt(0);
    __syncthreads();

    // S = Q @ K^T for this wave's 16x64 strip
    v8f sv[4];
#pragma unroll
    for (int nt = 0; nt < 4; ++nt) {
      v8f c = vz;
#pragma unroll
      for (int kb = 0; kb < 4; ++kb) {
        const __bf16* bb = Ks + (nt * 16 + m) * 136 + kb * 32 + 16 * half;
        v16bf bf = frag2(bb, bb + 8);
        c = __builtin_amdgcn_wmma_f32_16x16x32_bf16(false, qf[kb], false, bf,
                                                    (short)0, c, false, false);
      }
      sv[nt] = c;
    }
    // scale + mask (key index = nt*16 + m for this lane)
#pragma unroll
    for (int nt = 0; nt < 4; ++nt) {
      const bool valid = smask[nt * 16 + m] != 0;
#pragma unroll
      for (int r = 0; r < 8; ++r) {
        float s = sv[nt][r] * scale;
        sv[nt][r] = valid ? s : -3.0e38f;
      }
    }
    // online softmax: row r+8*half lives entirely in this 16-lane half at slot r
#pragma unroll
    for (int r = 0; r < 8; ++r) {
      float rmax = fmaxf(fmaxf(sv[0][r], sv[1][r]), fmaxf(sv[2][r], sv[3][r]));
#pragma unroll
      for (int off = 1; off < 16; off <<= 1) rmax = fmaxf(rmax, __shfl_xor(rmax, off, 32));
      const float mnew = fmaxf(mrow[r], rmax);
      const float alpha = __expf(mrow[r] - mnew);
      float rsum = 0.f;
#pragma unroll
      for (int nt = 0; nt < 4; ++nt) {
        const float p = __expf(sv[nt][r] - mnew);
        sv[nt][r] = p;
        rsum += p;
      }
#pragma unroll
      for (int off = 1; off < 16; off <<= 1) rsum += __shfl_xor(rsum, off, 32);
      lrow[r] = lrow[r] * alpha + rsum;
      mrow[r] = mnew;
#pragma unroll
      for (int nt = 0; nt < 8; ++nt) acc[nt][r] *= alpha;
    }
    // write P as bf16 into A-fragment-friendly row-major layout
#pragma unroll
    for (int nt = 0; nt < 4; ++nt)
#pragma unroll
      for (int r = 0; r < 8; ++r)
        Pb[(w * 16 + r + 8 * half) * 72 + nt * 16 + m] = (__bf16)sv[nt][r];
    __syncthreads();

    // ctx += P @ V
#pragma unroll
    for (int kb = 0; kb < 2; ++kb) {
      const __bf16* pa = Pb + (w * 16 + m) * 72 + kb * 32;
      v16bf af = frag2(pa + 16 * half, pa + 16 * half + 8);
#pragma unroll
      for (int nt = 0; nt < 8; ++nt) {
        const __bf16* vb = Vt + (nt * 16 + m) * 72 + kb * 32 + 16 * half;
        v16bf bf = frag2(vb, vb + 8);
        acc[nt] = __builtin_amdgcn_wmma_f32_16x16x32_bf16(false, af, false, bf,
                                                          (short)0, acc[nt], false, false);
      }
    }
    __syncthreads();
  }

  // epilogue: out = ctx / l
  const int qrow0 = q0 + w * 16 + 8 * half;
  float* ob = out + ((size_t)(b * MQc + qrow0)) * HD + h * Dc + m;
#pragma unroll
  for (int nt = 0; nt < 8; ++nt) {
    const float invl0 = 1.0f;  // per-row divide below
#pragma unroll
    for (int r = 0; r < 8; ++r)
      ob[(size_t)r * HD + nt * 16] = acc[nt][r] / lrow[r];
    (void)invl0;
  }
}

// ---------------------------------------------------------------- launcher
extern "C" void kernel_launch(void* const* d_in, const int* in_sizes, int n_in,
                              void* d_out, int out_size, void* d_ws, size_t ws_size,
                              hipStream_t stream) {
  constexpr int B = 8, MQ = 512, MK = 2048, Q_DIM = 1024, KV_DIM = 256;
  constexpr int QKC = 1024;

  const float* hidden = (const float*)d_in[0];
  const float* inputs = (const float*)d_in[1];
  const int*   maskp  = (const int*)d_in[2];
  const float* ln1w   = (const float*)d_in[3];
  const float* ln1b   = (const float*)d_in[4];
  const float* ln2w   = (const float*)d_in[5];
  const float* ln2b   = (const float*)d_in[6];
  const float* Wq     = (const float*)d_in[7];
  const float* bq     = (const float*)d_in[8];
  const float* Wk     = (const float*)d_in[9];
  const float* bk     = (const float*)d_in[10];
  const float* Wv     = (const float*)d_in[11];
  const float* bv     = (const float*)d_in[12];

  char* ws = (char*)d_ws;
  __bf16* hsb = (__bf16*)(ws + 0);          // 8 MB   [B*MQ, Q_DIM]
  __bf16* kvb = (__bf16*)(ws + 8388608);    // 8 MB   [B*MK, KV_DIM]
  __bf16* Wqb = (__bf16*)(ws + 16777216);   // 2 MB
  __bf16* Wkb = (__bf16*)(ws + 18874368);   // 0.5 MB
  __bf16* Wvb = (__bf16*)(ws + 19398656);   // 0.5 MB
  __bf16* Qb  = (__bf16*)(ws + 19922944);   // 8 MB   [B*MQ, 1024]
  __bf16* Kb  = (__bf16*)(ws + 28311552);   // 32 MB  [B*MK, 1024]
  __bf16* Vb  = (__bf16*)(ws + 61865984);   // 32 MB  [B*MK, 1024]

  ln_cast_kernel<<<B * MQ, 256, 0, stream>>>(hidden, ln1w, ln1b, hsb, Q_DIM);
  ln_cast_kernel<<<B * MK, 256, 0, stream>>>(inputs, ln2w, ln2b, kvb, KV_DIM);

  cast_bf16_kernel<<<(Q_DIM * QKC) / 256, 256, 0, stream>>>(Wq, Wqb, Q_DIM * QKC);
  cast_bf16_kernel<<<(KV_DIM * QKC) / 256, 256, 0, stream>>>(Wk, Wkb, KV_DIM * QKC);
  cast_bf16_kernel<<<(KV_DIM * QKC) / 256, 256, 0, stream>>>(Wv, Wvb, KV_DIM * QKC);

  dim3 gq(QKC / 128, (B * MQ) / 64);
  gemm_bf16_kernel<Q_DIM, QKC><<<gq, 256, 0, stream>>>(hsb, Wqb, bq, Qb);
  dim3 gkv(QKC / 128, (B * MK) / 64);
  gemm_bf16_kernel<KV_DIM, QKC><<<gkv, 256, 0, stream>>>(kvb, Wkb, bk, Kb);
  gemm_bf16_kernel<KV_DIM, QKC><<<gkv, 256, 0, stream>>>(kvb, Wvb, bv, Vb);

  flash_attn_kernel<<<B * 8 * (MQ / 64), 128, 0, stream>>>(Qb, Kb, Vb, maskp, (float*)d_out);
}
